// ProSurv_46462956208617
// MI455X (gfx1250) — matrix operand order repaired and checked by reading, not compiled, not run on hardware
//
#include <hip/hip_runtime.h>

typedef __attribute__((ext_vector_type(16))) _Float16 v16h;
typedef __attribute__((ext_vector_type(8)))  float    v8f;
typedef __attribute__((ext_vector_type(4)))  float    f4;

#define EPSN 1e-12f

// ---------------------------------------------------------------------------
// Kernel 1: p[bank][c][d] = (1/512) * sum_m normalize(bank[c][m][:])[d]
// grid.x = 8  (bank*4 + class), block = 256 threads (8 waves of 32)
// Each wave normalizes 64 rows (D=256, 8 floats per lane, coalesced) and
// accumulates the class mean in registers; one atomicAdd burst at the end.
// ---------------------------------------------------------------------------
__global__ void proto_mean_kernel(const float* __restrict__ bank_path,
                                  const float* __restrict__ bank_geno,
                                  float* __restrict__ proto /* [2][4][256] */)
{
    const int bc     = blockIdx.x;        // 0..7
    const int bankid = bc >> 2;
    const int cls    = bc & 3;
    const float* bank = bankid ? bank_geno : bank_path;
    const int wave = threadIdx.x >> 5;    // 0..7
    const int lane = threadIdx.x & 31;

    float acc[8];
#pragma unroll
    for (int j = 0; j < 8; ++j) acc[j] = 0.f;

    for (int m = wave; m < 512; m += 8) {
        const float* row = bank + ((size_t)cls * 512 + (size_t)m) * 256;
        float x[8];
        float ss = 0.f;
#pragma unroll
        for (int j = 0; j < 8; ++j) {
            x[j] = row[j * 32 + lane];    // coalesced 128B per j
            ss += x[j] * x[j];
        }
#pragma unroll
        for (int off = 16; off >= 1; off >>= 1)
            ss += __shfl_xor(ss, off, 32);
        const float inv = 1.f / fmaxf(sqrtf(ss), EPSN);
#pragma unroll
        for (int j = 0; j < 8; ++j) acc[j] += x[j] * inv;
    }

    float* dst = proto + ((size_t)bankid * 4 + (size_t)cls) * 256;
#pragma unroll
    for (int j = 0; j < 8; ++j)
        atomicAdd(&dst[j * 32 + lane], acc[j] * (1.0f / 512.0f));
}

// ---------------------------------------------------------------------------
// Kernel 2: one wave = one tile of 16 samples of one modality.
//   cm[16 rows][4 classes] via 8x v_wmma_f32_16x16x32_f16 (K = 256),
//   post-scaled by 1/max(||h_row||, eps), then per-sample survival loss,
//   wave-reduced and atomically added into the scalar output.
// grid = (128, 2), block = 256 (8 waves) -> 1024 tiles * 16 = 16384 samples.
// ---------------------------------------------------------------------------
__global__ void cm_loss_kernel(const float* __restrict__ h_path,
                               const float* __restrict__ h_geno,
                               const float* __restrict__ proto,
                               const int*   __restrict__ label,
                               const int*   __restrict__ censor,
                               float* __restrict__ out)
{
    const int wave = threadIdx.x >> 5;
    const int lane = threadIdx.x & 31;
    const int mod  = blockIdx.y;                   // 0 = path, 1 = geno
    const int tile = blockIdx.x * 8 + wave;        // 0..1023
    const int row  = lane & 15;                    // sample-in-tile / column id
    const int hi   = lane >> 4;                    // lane half (K split)

    const float* h = mod ? h_geno : h_path;
    const float* p = proto + (size_t)mod * 1024;   // [4][256]

    // ---- B fragments: column (lane&15) = class (zeros for class >= 4) ----
    // B layout as loaded here (16-bit, 32x16): lanes 0-15 hold K = 32s..32s+15 of column
    // `lane`; lanes 16-31 hold K = 32s+16..32s+31 of column `lane-16`.
    v16h bfrag[8];
#pragma unroll
    for (int s = 0; s < 8; ++s) {
        const int kbase = s * 32 + hi * 16;
#pragma unroll
        for (int e = 0; e < 16; ++e) bfrag[s][e] = (_Float16)0.f;
        if (row < 4) {
            const float* pc = p + (size_t)row * 256 + kbase;
#pragma unroll
            for (int e = 0; e < 16; ++e) bfrag[s][e] = (_Float16)pc[e];
        }
    }

    // ---- A fragments + WMMA over K; fp32 sum-of-squares on the side ----
    // A layout (16-bit, 16x32): lane l<16 holds row l, K {32s+0..15} in
    // elems 0..15; lane l+16 holds K {32s+16..31} (matches B's K order).
    // Two lanes cover a full row.
    const float* hrow = h + ((size_t)tile * 16 + (size_t)row) * 256;
    v8f acc = {};
    float ss = 0.f;
#pragma unroll
    for (int s = 0; s < 8; ++s) {
        const int o1 = s * 32 + hi * 16;
        f4 a0 = *(const f4*)(hrow + o1);
        f4 a1 = *(const f4*)(hrow + o1 + 4);
        f4 a2 = *(const f4*)(hrow + o1 + 8);
        f4 a3 = *(const f4*)(hrow + o1 + 12);
        v16h afrag;
#pragma unroll
        for (int e = 0; e < 4; ++e) {
            afrag[e]      = (_Float16)a0[e];
            afrag[e + 4]  = (_Float16)a1[e];
            afrag[e + 8]  = (_Float16)a2[e];
            afrag[e + 12] = (_Float16)a3[e];
            ss += a0[e] * a0[e] + a1[e] * a1[e] + a2[e] * a2[e] + a3[e] * a3[e];
        }
        // D = A*B + C : v_wmma_f32_16x16x32_f16
        acc = __builtin_amdgcn_wmma_f32_16x16x32_f16(false, afrag, false, bfrag[s],
                                                     (short)0, acc, false, false);
    }

    // ---- row inverse norms (lanes l and l+16 jointly cover row l&15) ----
    ss += __shfl_xor(ss, 16, 32);
    const float inv = 1.f / fmaxf(sqrtf(ss), EPSN);

    // ---- scale D rows, stash cm[row][class] in LDS ----
    // D layout: acc[j] -> row j (lanes 0-15, col=lane) / row j+8 (lanes 16-31).
    __shared__ float lds_cm[8][16][4];
#pragma unroll
    for (int j = 0; j < 8; ++j) {
        const int r   = j + hi * 8;
        const float invj = __shfl(inv, r, 32);     // lane r holds row r's inv
        const float cm   = acc[j] * invj;
        if (row < 4) lds_cm[wave][r][row] = cm;
    }
    __syncthreads();

    // ---- per-sample loss (lanes 0..15 handle the 16 samples) ----
    float contrib = 0.f;
    if (hi == 0) {
        float sv[4];
#pragma unroll
        for (int c = 0; c < 4; ++c) sv[c] = lds_cm[wave][row][c];
        const float sum = sv[0] + sv[1] + sv[2] + sv[3];
        const int gid = tile * 16 + row;
        const int lab = label[gid];
        const int cen = censor[gid];
        float pos, neg;
        if (cen == 0) {
            pos = sv[lab];
            neg = (sum - sv[lab]) * (1.0f / 3.0f);
        } else {
            // general censored formula also covers lab==0 (pos=sum/4, neg=0)
            float sge = 0.f;
#pragma unroll
            for (int c = 0; c < 4; ++c) if (c >= lab) sge += sv[c];
            pos = sge / (float)(4 - lab);
            neg = (sum - sge) / (float)(lab > 0 ? lab : 1);
        }
        contrib = neg - pos;
    }
#pragma unroll
    for (int off = 16; off >= 1; off >>= 1)
        contrib += __shfl_xor(contrib, off, 32);
    if (lane == 0) atomicAdd(out, contrib);
}

// ---------------------------------------------------------------------------
extern "C" void kernel_launch(void* const* d_in, const int* in_sizes, int n_in,
                              void* d_out, int out_size, void* d_ws, size_t ws_size,
                              hipStream_t stream) {
    const float* h_path = (const float*)d_in[0];
    const float* h_geno = (const float*)d_in[1];
    const float* bank_p = (const float*)d_in[2];
    const float* bank_g = (const float*)d_in[3];
    const int*   label  = (const int*)d_in[4];
    const int*   censor = (const int*)d_in[5];
    float* out   = (float*)d_out;
    float* proto = (float*)d_ws;                   // 2*4*256 floats = 8 KB

    hipMemsetAsync(proto, 0, 2048 * sizeof(float), stream);
    hipMemsetAsync(out,   0, sizeof(float), stream);

    proto_mean_kernel<<<dim3(8, 1, 1), 256, 0, stream>>>(bank_p, bank_g, proto);
    cm_loss_kernel<<<dim3(128, 2, 1), 256, 0, stream>>>(h_path, h_geno, proto,
                                                        label, censor, out);
}
